// MultiScaleVQ_61701500175140
// MI455X (gfx1250) — compile-verified, matrix operand order checked by reading, not run on hardware
//
#include <hip/hip_runtime.h>
#include <hip/hip_bf16.h>
#include <cstdint>
#include <cstddef>

typedef __attribute__((ext_vector_type(16))) _Float16     v16h;
typedef __attribute__((ext_vector_type(8)))  float        v8f;
typedef __attribute__((ext_vector_type(4)))  unsigned int u32x4;
typedef __attribute__((ext_vector_type(8)))  unsigned int u32x8;

#define DCH    32
#define KCB    4096
#define BATCH  64
#define FULLHW 32
#define SAMPLE_STRIDE 32768   // 32 ch * 32 * 32 floats, max per-sample plane
#define QWAVES 16             // waves per quantize workgroup (share codebook in LDS)
#define CB_LDS_BYTES (KCB * DCH * 2)   // 256 KB packed f16 codebook

// ---------------------------------------------------------------------------
// Codebook pack: fp32 (4096,32) -> f16 WMMA B-tiles + fp32 ||c||^2
// B tile t (16 codes): lane l holds code 16t+(l&15), feats (l<16 ? 0..15 : 16..31)
// ---------------------------------------------------------------------------
__global__ void pack_codebook_kernel(const float* __restrict__ cb,
                                     _Float16* __restrict__ cbPack,
                                     float* __restrict__ cnorm) {
  const int t    = blockIdx.x;       // 0..255
  const int lane = threadIdx.x;      // 0..31
  const int col  = lane & 15;
  const int half = lane >> 4;
  const int code = t * 16 + col;
  const float* src = cb + (size_t)code * DCH + half * 16;
  _Float16* dst = cbPack + ((size_t)t * 32 + lane) * 16;
#pragma unroll
  for (int j = 0; j < 16; ++j) dst[j] = (_Float16)src[j];
  if (half == 0) {
    float s = 0.f;
#pragma unroll
    for (int f = 0; f < DCH; ++f) {
      float v = cb[(size_t)code * DCH + f];
      s = __builtin_fmaf(v, v, s);
    }
    cnorm[code] = s;
  }
}

// ---------------------------------------------------------------------------
// TDM: stage a 1-D run of 8-byte elements from global into LDS.
// D# group0: {count=1, lds_addr, global_addr[31:0], global_addr[56:32]|type=2}
// D# group1: data_size=3 (8B), tensor_dim0=n, tile_dim0=n, dim0_stride=n.
// 2-operand form (VADDR2/VADDR3 = NULL -> <=2D tensor).  Tracked by TENSORcnt.
// ---------------------------------------------------------------------------
__device__ __forceinline__ void tdm_load_to_lds(unsigned int lds_off,
                                                const void* gsrc,
                                                unsigned int n_elem8) {
  unsigned long long ga = (unsigned long long)(uintptr_t)gsrc;
  u32x4 g0;
  g0[0] = 1u;                                        // count=1 (user descriptor)
  g0[1] = lds_off;                                   // lds_addr (bytes)
  g0[2] = (unsigned int)ga;                          // global_addr[31:0]
  g0[3] = (unsigned int)(ga >> 32) | 0x80000000u;    // global_addr[56:32] | type=2
  u32x8 g1;
  g1[0] = 3u << 16;                                  // data_size = 8 bytes
  g1[1] = (n_elem8 & 0xFFFFu) << 16;                 // tensor_dim0[15:0]
  g1[2] = ((n_elem8 >> 16) & 0xFFFFu) | (1u << 16);  // tensor_dim0[31:16], tensor_dim1=1
  g1[3] = (n_elem8 & 0xFFFFu) << 16;                 // tile_dim0 (bits 127:112)
  g1[4] = 0u;                                        // tile_dim1=0, tile_dim2=0
  g1[5] = n_elem8;                                   // tensor_dim0_stride[31:0]
  g1[6] = 0u;
  g1[7] = 0u;
  asm volatile("tensor_load_to_lds %0, %1" :: "s"(g0), "s"(g1) : "memory");
}

// ---------------------------------------------------------------------------
// Quantize: 16 waves per workgroup.  TDM preloads the entire packed codebook
// (256 KB f16) into LDS once; each wave owns one 16-position tile, holds its
// A fragment (16x32 f16 residuals) in registers for the whole kernel, and
// loops the 256 codebook B-tiles out of LDS.  v_wmma_f32_16x16x32_f16 does
// the FULL D=32 reduction per 16x16 tile of dot products; unrolled x2 with
// independent accumulators so the second WMMA hides the WMMA->VALU hazard of
// the first.  dist = ||c||^2 - 2*dot, running argmin, shfl_xor cross-column.
// ---------------------------------------------------------------------------
__global__ void quantize_kernel(const float* __restrict__ zf,        // (B,N,32)
                                const _Float16* __restrict__ cbPack,
                                const float* __restrict__ cnorm,
                                int* __restrict__ idx_ws,            // (B,N)
                                float* __restrict__ idx_outf,        // (B,N)
                                int N, int tilesPer) {
  extern __shared__ _Float16 cbLds[];                // KCB*DCH f16 = 256 KB
  const int lane = threadIdx.x & 31;
  const int wave = threadIdx.x >> 5;
  const int col  = lane & 15;
  const int half = lane >> 4;

  if (wave == 0) {
    tdm_load_to_lds((unsigned int)(uintptr_t)cbLds, cbPack, KCB * DCH / 4);
    __builtin_amdgcn_s_wait_tensorcnt(0);
  }
  __syncthreads();

  const int gtile = blockIdx.x * QWAVES + wave;
  const int b     = gtile / tilesPer;
  const int ptile = gtile - b * tilesPer;
  if (b >= BATCH) return;                            // whole-wave uniform exit
  const int posBase = ptile * 16;

  // A-matrix fragment: row = col (position in tile), feats half*16..+15
  int p  = posBase + col;
  int pc = p < N ? p : N - 1;                        // clamp padding rows
  const float* asrc = zf + ((size_t)b * N + pc) * DCH + half * 16;
  v16h a;
#pragma unroll
  for (int j = 0; j < 16; ++j) a[j] = (_Float16)asrc[j];

  float best[8];
  int   bidx[8];
#pragma unroll
  for (int r = 0; r < 8; ++r) { best[r] = 3.4e38f; bidx[r] = 0; }

  for (int t = 0; t < KCB / 16; t += 2) {
    const v16h* bp0 = (const v16h*)(cbLds + ((size_t)t * 32 + lane) * 16);
    const v16h* bp1 = (const v16h*)(cbLds + ((size_t)(t + 1) * 32 + lane) * 16);
    v16h bm0 = *bp0;
    v16h bm1 = *bp1;
    v8f c0 = {0.f, 0.f, 0.f, 0.f, 0.f, 0.f, 0.f, 0.f};
    v8f c1 = {0.f, 0.f, 0.f, 0.f, 0.f, 0.f, 0.f, 0.f};
    c0 = __builtin_amdgcn_wmma_f32_16x16x32_f16(false, a, false, bm0, (short)0, c0, false, false);
    c1 = __builtin_amdgcn_wmma_f32_16x16x32_f16(false, a, false, bm1, (short)0, c1, false, false);
    const float cn0 = cnorm[t * 16 + col];
    const float cn1 = cnorm[t * 16 + 16 + col];
    const int code0 = t * 16 + col;
    const int code1 = code0 + 16;
#pragma unroll
    for (int r = 0; r < 8; ++r) {
      float d0 = __builtin_fmaf(-2.f, c0[r], cn0);
      if (d0 < best[r]) { best[r] = d0; bidx[r] = code0; }
    }
#pragma unroll
    for (int r = 0; r < 8; ++r) {
      float d1 = __builtin_fmaf(-2.f, c1[r], cn1);
      if (d1 < best[r]) { best[r] = d1; bidx[r] = code1; }
    }
  }

  // argmin across 16 columns (lanes 0-15 share rows 0-7; 16-31 share rows 8-15)
#pragma unroll
  for (int m = 1; m < 16; m <<= 1) {
#pragma unroll
    for (int r = 0; r < 8; ++r) {
      float ob = __shfl_xor(best[r], m, 32);
      int   oi = __shfl_xor(bidx[r], m, 32);
      if (ob < best[r] || (ob == best[r] && oi < bidx[r])) { best[r] = ob; bidx[r] = oi; }
    }
  }

  if (col == 0) {
#pragma unroll
    for (int r = 0; r < 8; ++r) {
      int row = r + half * 8;
      int pp  = posBase + row;
      if (pp < N) {
        idx_ws  [(size_t)b * N + pp] = bidx[r];
        idx_outf[(size_t)b * N + pp] = (float)bidx[r];
      }
    }
  }
}

// ---------------------------------------------------------------------------
// Bicubic (Keys a = -0.5, half-pixel centers, edge clamp)
// ---------------------------------------------------------------------------
__device__ __forceinline__ float cubicw(float t) {
  float a = fabsf(t);
  if (a <= 1.f) return ((1.5f * a - 2.5f) * a) * a + 1.f;
  if (a < 2.f)  return (((-0.5f * a + 2.5f) * a) - 4.f) * a + 2.f;
  return 0.f;
}

template <int IN, int OUT>
__device__ __forceinline__ float bicubic_at(const float* __restrict__ ch, int oy, int ox) {
  const float scale = (float)IN / (float)OUT;
  float fy = ((float)oy + 0.5f) * scale - 0.5f;
  float fx = ((float)ox + 0.5f) * scale - 0.5f;
  int iy = (int)floorf(fy), ix = (int)floorf(fx);
  float ty = fy - (float)iy, tx = fx - (float)ix;
  float acc = 0.f;
#pragma unroll
  for (int ky = -1; ky <= 2; ++ky) {
    int y = iy + ky; y = y < 0 ? 0 : (y > IN - 1 ? IN - 1 : y);
    float wy = cubicw(ty - (float)ky);
#pragma unroll
    for (int kx = -1; kx <= 2; ++kx) {
      int x = ix + kx; x = x < 0 ? 0 : (x > IN - 1 ? IN - 1 : x);
      float wx = cubicw(tx - (float)kx);
      acc = __builtin_fmaf(wy * wx, ch[y * IN + x], acc);
    }
  }
  return acc;
}

// ---------------------------------------------------------------------------
// residual: f_hat_cur = bicubic_up(f_hat_prev); res = area_down(z_e) - f_hat_cur
// ---------------------------------------------------------------------------
template <int S, int SPREV>
__global__ void residual_kernel(const float* __restrict__ z_e,
                                const float* __restrict__ fhat_prev,
                                float* __restrict__ fhat_cur,
                                float* __restrict__ zf_out) {
  const int b  = blockIdx.x;
  const int N  = S * S;
  const int BS = FULLHW / S;
  const float* zb = z_e + (size_t)b * SAMPLE_STRIDE;
  const float* fp = fhat_prev + (size_t)b * SAMPLE_STRIDE;
  float*       fc = fhat_cur + (size_t)b * SAMPLE_STRIDE;
  for (int e = threadIdx.x; e < DCH * N; e += blockDim.x) {
    int d = e / N, p = e % N;
    int y = p / S, x = p % S;
    const float* zc = zb + (size_t)d * FULLHW * FULLHW;
    float dn = 0.f;
    for (int i = 0; i < BS; ++i)
      for (int j = 0; j < BS; ++j)
        dn += zc[(y * BS + i) * FULLHW + (x * BS + j)];
    dn *= 1.f / (float)(BS * BS);
    float up = 0.f;
    if constexpr (SPREV > 0)
      up = bicubic_at<SPREV, S>(fp + (size_t)d * SPREV * SPREV, y, x);
    fc[(d * S + y) * S + x] = up;
    zf_out[((size_t)b * N + p) * DCH + d] = dn - up;
  }
}

// ---------------------------------------------------------------------------
// gather codebook rows into (D,S,S) conv layout + commit = mean((res-zq)^2)
// ---------------------------------------------------------------------------
template <int S>
__global__ void gather_commit_kernel(const int* __restrict__ idx_ws,
                                     const float* __restrict__ zf,
                                     const float* __restrict__ cb,
                                     float* __restrict__ zq_ws,
                                     float* __restrict__ commit) {
  const int b = blockIdx.x;
  const int N = S * S;
  __shared__ float red[256];
  float acc = 0.f;
  for (int e = threadIdx.x; e < N * DCH; e += blockDim.x) {
    int p = e / DCH, d = e % DCH;
    int y = p / S, x = p % S;
    int id  = idx_ws[(size_t)b * N + p];
    float q = cb[(size_t)id * DCH + d];
    float z = zf[((size_t)b * N + p) * DCH + d];
    zq_ws[(size_t)b * SAMPLE_STRIDE + (d * S + y) * S + x] = q;
    float df = z - q;
    acc = __builtin_fmaf(df, df, acc);
  }
  red[threadIdx.x] = acc;
  __syncthreads();
  for (int st = blockDim.x / 2; st > 0; st >>= 1) {
    if ((int)threadIdx.x < st) red[threadIdx.x] += red[threadIdx.x + st];
    __syncthreads();
  }
  if (threadIdx.x == 0) atomicAdd(&commit[b], red[0] / (float)(N * DCH));
}

// ---------------------------------------------------------------------------
// direct 3x3 conv, D=32 in/out, zero padding.  accumulate=1 -> out += conv.
// ---------------------------------------------------------------------------
template <int S>
__global__ void conv3x3_kernel(const float* __restrict__ in,
                               const float* __restrict__ w,
                               const float* __restrict__ bias,
                               float* __restrict__ out,
                               size_t out_sample_stride,
                               int accumulate,
                               float* __restrict__ out2) {
  const int b = blockIdx.x;
  const float* ib = in + (size_t)b * SAMPLE_STRIDE;
  float*       ob = out + (size_t)b * out_sample_stride;
  for (int e = threadIdx.x + blockIdx.y * blockDim.x; e < DCH * S * S;
       e += blockDim.x * gridDim.y) {
    int dout = e / (S * S), p = e % (S * S);
    int y = p / S, x = p % S;
    float acc = bias[dout];
    for (int di = 0; di < DCH; ++di) {
      const float* ic = ib + (size_t)di * S * S;
      const float* wc = w + ((size_t)dout * DCH + di) * 9;
#pragma unroll
      for (int ky = 0; ky < 3; ++ky) {
        int yy = y + ky - 1;
        if (yy < 0 || yy >= S) continue;
#pragma unroll
        for (int kx = 0; kx < 3; ++kx) {
          int xx = x + kx - 1;
          if (xx < 0 || xx >= S) continue;
          acc = __builtin_fmaf(ic[yy * S + xx], wc[ky * 3 + kx], acc);
        }
      }
    }
    int o = (dout * S + y) * S + x;
    if (accumulate) ob[o] += acc; else ob[o] = acc;
    if (out2) out2[(size_t)b * SAMPLE_STRIDE + o] = acc;
  }
}

// ---------------------------------------------------------------------------
// bicubic upsample f_hat (S x S) -> (32 x 32) staging for the post conv
// ---------------------------------------------------------------------------
template <int S>
__global__ void upsample32_kernel(const float* __restrict__ fhat,
                                  float* __restrict__ tmp) {
  const int b = blockIdx.x;
  const float* fb = fhat + (size_t)b * SAMPLE_STRIDE;
  float*       tb = tmp + (size_t)b * SAMPLE_STRIDE;
  for (int e = threadIdx.x + blockIdx.y * blockDim.x; e < DCH * FULLHW * FULLHW;
       e += blockDim.x * gridDim.y) {
    int d = e / (FULLHW * FULLHW), p = e % (FULLHW * FULLHW);
    int y = p >> 5, x = p & 31;
    tb[e] = bicubic_at<S, FULLHW>(fb + (size_t)d * S * S, y, x);
  }
}

__global__ void zero_kernel(float* __restrict__ p, int n) {
  int i = blockIdx.x * blockDim.x + threadIdx.x;
  if (i < n) p[i] = 0.f;
}

// ---------------------------------------------------------------------------
// per-scale driver
// ---------------------------------------------------------------------------
template <int K, int S, int SPREV>
static void run_scale(const float* z_e, const float* cb,
                      const float* phi_w, const float* phi_b,
                      const float* post_w, const float* post_b,
                      const _Float16* cbPack, const float* cnorm, int* idx_ws,
                      float* fhatA, float* fhatB, float* zq_ws, float* tmp,
                      float* out, size_t offFinal, size_t offIdx, size_t offPart,
                      size_t offCommit, size_t offZf, hipStream_t stream) {
  constexpr int N = S * S;
  constexpr int prefix[6] = {0, 1, 5, 21, 85, 341};
  float* fhat_cur  = (K % 2 == 0) ? fhatA : fhatB;
  float* fhat_prev = (K % 2 == 0) ? fhatB : fhatA;
  float* zf_k   = out + offZf + (size_t)BATCH * prefix[K] * DCH;
  float* idxf_k = out + offIdx + (size_t)BATCH * prefix[K];
  int*   idxw_k = idx_ws + (size_t)BATCH * prefix[K];
  float* part_k = out + offPart + (size_t)K * BATCH * SAMPLE_STRIDE;

  residual_kernel<S, SPREV><<<dim3(BATCH), dim3(256), 0, stream>>>(
      z_e, fhat_prev, fhat_cur, zf_k);

  const int tilesPer = (N + 15) / 16;
  const int nblocks  = (BATCH * tilesPer + QWAVES - 1) / QWAVES;
  quantize_kernel<<<dim3(nblocks), dim3(QWAVES * 32), CB_LDS_BYTES, stream>>>(
      zf_k, cbPack, cnorm, idxw_k, idxf_k, N, tilesPer);

  gather_commit_kernel<S><<<dim3(BATCH), dim3(256), 0, stream>>>(
      idxw_k, zf_k, cb, zq_ws, out + offCommit);
  conv3x3_kernel<S><<<dim3(BATCH, (S >= 16 ? 8 : 1)), dim3(256), 0, stream>>>(
      zq_ws, phi_w + (size_t)K * DCH * DCH * 9, phi_b + (size_t)K * DCH,
      fhat_cur, SAMPLE_STRIDE, /*accumulate=*/1, nullptr);
  upsample32_kernel<S><<<dim3(BATCH, 8), dim3(256), 0, stream>>>(fhat_cur, tmp);
  conv3x3_kernel<FULLHW><<<dim3(BATCH, 8), dim3(256), 0, stream>>>(
      tmp, post_w, post_b, part_k, SAMPLE_STRIDE, /*accumulate=*/0,
      (K == 5) ? (out + offFinal) : nullptr);
}

static inline size_t align256(size_t v) { return (v + 255) & ~(size_t)255; }

extern "C" void kernel_launch(void* const* d_in, const int* in_sizes, int n_in,
                              void* d_out, int out_size, void* d_ws, size_t ws_size,
                              hipStream_t stream) {
  const float* z_e    = (const float*)d_in[0];   // (64,32,32,32)
  const float* cb     = (const float*)d_in[1];   // (4096,32)
  const float* phi_w  = (const float*)d_in[2];   // (6,32,32,3,3)
  const float* phi_b  = (const float*)d_in[3];   // (6,32)
  const float* post_w = (const float*)d_in[4];   // (32,32,3,3)
  const float* post_b = (const float*)d_in[5];   // (32,)
  float* out = (float*)d_out;

  // workspace carve-up
  char* ws = (char*)d_ws;
  _Float16* cbPack = (_Float16*)ws; ws += align256((size_t)KCB * DCH * sizeof(_Float16));
  float* cnorm = (float*)ws;        ws += align256((size_t)KCB * sizeof(float));
  int* idx_ws  = (int*)ws;          ws += align256((size_t)BATCH * 1365 * sizeof(int));
  float* fhatA = (float*)ws;        ws += align256((size_t)BATCH * SAMPLE_STRIDE * sizeof(float));
  float* fhatB = (float*)ws;        ws += align256((size_t)BATCH * SAMPLE_STRIDE * sizeof(float));
  float* zq_ws = (float*)ws;        ws += align256((size_t)BATCH * SAMPLE_STRIDE * sizeof(float));
  float* tmp   = (float*)ws;        ws += align256((size_t)BATCH * SAMPLE_STRIDE * sizeof(float));

  // output region offsets (floats), in reference return order
  const size_t offFinal  = 0;                                            // (B,32,32,32)
  const size_t offIdx    = (size_t)BATCH * SAMPLE_STRIDE;                // 6x (B,s^2)
  const size_t offPart   = offIdx + (size_t)BATCH * 1365;                // 6x (B,32,32,32)
  const size_t offCommit = offPart + (size_t)6 * BATCH * SAMPLE_STRIDE;  // (B,)
  const size_t offZf     = offCommit + BATCH;                            // 6x (B,s^2,32)

  pack_codebook_kernel<<<dim3(KCB / 16), dim3(32), 0, stream>>>(cb, cbPack, cnorm);
  zero_kernel<<<dim3(1), dim3(64), 0, stream>>>(out + offCommit, BATCH);

  run_scale<0, 1, 0 >(z_e, cb, phi_w, phi_b, post_w, post_b, cbPack, cnorm, idx_ws,
                      fhatA, fhatB, zq_ws, tmp, out, offFinal, offIdx, offPart, offCommit, offZf, stream);
  run_scale<1, 2, 1 >(z_e, cb, phi_w, phi_b, post_w, post_b, cbPack, cnorm, idx_ws,
                      fhatA, fhatB, zq_ws, tmp, out, offFinal, offIdx, offPart, offCommit, offZf, stream);
  run_scale<2, 4, 2 >(z_e, cb, phi_w, phi_b, post_w, post_b, cbPack, cnorm, idx_ws,
                      fhatA, fhatB, zq_ws, tmp, out, offFinal, offIdx, offPart, offCommit, offZf, stream);
  run_scale<3, 8, 4 >(z_e, cb, phi_w, phi_b, post_w, post_b, cbPack, cnorm, idx_ws,
                      fhatA, fhatB, zq_ws, tmp, out, offFinal, offIdx, offPart, offCommit, offZf, stream);
  run_scale<4, 16, 8>(z_e, cb, phi_w, phi_b, post_w, post_b, cbPack, cnorm, idx_ws,
                      fhatA, fhatB, zq_ws, tmp, out, offFinal, offIdx, offPart, offCommit, offZf, stream);
  run_scale<5, 32, 16>(z_e, cb, phi_w, phi_b, post_w, post_b, cbPack, cnorm, idx_ws,
                      fhatA, fhatB, zq_ws, tmp, out, offFinal, offIdx, offPart, offCommit, offZf, stream);
}
